// Att_MIL_Pooling_16844861735559
// MI455X (gfx1250) — matrix operand order repaired and checked by reading, not compiled, not run on hardware
//
#include <hip/hip_runtime.h>
#include <hip/hip_bf16.h>
#include <cmath>

typedef __attribute__((ext_vector_type(16))) __bf16 v16bf;
typedef __attribute__((ext_vector_type(8)))  float  v8f;

#define NB 256                 // bags
#define SZB 64                 // bag size
#define NN (NB * SZB)          // 16384 instances
#define LL 64                  // L
#define HH 128                 // H
#define DD 256                 // D

// ---------------------------------------------------------------------------
// Kernel 1: m[row] = max_d inter_pre[row*DD + d], rows = N*L. One wave / row.
// Streams 1 GiB with float4 loads; wave32 xor-shuffle max reduction.
// ---------------------------------------------------------------------------
__global__ __launch_bounds__(256) void k_rowmax(const float* __restrict__ x,
                                                float* __restrict__ m) {
    const int wave = threadIdx.x >> 5;
    const int lane = threadIdx.x & 31;
    const long long row = (long long)blockIdx.x * 8 + wave;
    const float4* p = (const float4*)(x + row * DD);
    float4 a = p[lane * 2];
    float4 b = p[lane * 2 + 1];
    float mx = fmaxf(fmaxf(fmaxf(a.x, a.y), fmaxf(a.z, a.w)),
                     fmaxf(fmaxf(b.x, b.y), fmaxf(b.z, b.w)));
#pragma unroll
    for (int s = 16; s >= 1; s >>= 1)
        mx = fmaxf(mx, __shfl_xor(mx, s, 32));
    if (lane == 0) m[row] = mx;
}

// ---------------------------------------------------------------------------
// Kernel 2: attention logits via WMMA.
//   v = tanh(m @ Wv^T + bv), u = sigmoid(m @ Wu^T + bu)
//   att[n] = sum_h v*u*Ww[h] + bw
// Block = 256 threads (8 waves). Block covers 16 instances (M tile); wave w
// covers h-chunk [16w, 16w+16). Two bf16 WMMAs (K=32 each) per weight matrix.
// ---------------------------------------------------------------------------
__global__ __launch_bounds__(256) void k_att(const float* __restrict__ m,
                                             const float* __restrict__ Wv,
                                             const float* __restrict__ bvp,
                                             const float* __restrict__ Wu,
                                             const float* __restrict__ bup,
                                             const float* __restrict__ Ww,
                                             const float* __restrict__ bwp,
                                             float* __restrict__ att) {
    __shared__ float part[16];
    const int tid  = threadIdx.x;
    const int lane = tid & 31;
    const int wave = tid >> 5;
    const int half = lane >> 4;       // 0: lanes 0-15, 1: lanes 16-31
    const int col  = lane & 15;
    const int n0   = blockIdx.x * 16;

    if (tid < 16) part[tid] = 0.0f;
    __syncthreads();

    // A fragment: rows n0..n0+15 of m (NxL), K over L=64.
    // A k order chosen here, same as B (not the ISA table order): lane<16 holds K=0..15, lane>=16 K=16..31.
    const float* arow = m + (long long)(n0 + col) * LL;
    v16bf a0, a1;
#pragma unroll
    for (int j = 0; j < 16; ++j) {
        const int k = (half ? 16 : 0) + j;
        a0[j] = (__bf16)arow[k];
        a1[j] = (__bf16)arow[k + 32];
    }

    // B fragments: B[k][col] = W[h0+col][k]. This code's B k order (same as A): lanes 0-15 hold
    // K=0..15 (elem j -> K=j), lanes 16-31 hold K=16..31.
    const int h0 = wave * 16;
    const int h  = h0 + col;
    const int kb = half ? 16 : 0;
    const float* wvrow = Wv + (long long)h * LL;
    const float* wurow = Wu + (long long)h * LL;
    v16bf bv0, bv1, bu0, bu1;
#pragma unroll
    for (int j = 0; j < 16; ++j) {
        bv0[j] = (__bf16)wvrow[kb + j];
        bv1[j] = (__bf16)wvrow[kb + 32 + j];
        bu0[j] = (__bf16)wurow[kb + j];
        bu1[j] = (__bf16)wurow[kb + 32 + j];
    }

    v8f accv = {};
    v8f accu = {};
    accv = __builtin_amdgcn_wmma_f32_16x16x32_bf16(false, a0, false, bv0, (short)0, accv, false, false);
    accv = __builtin_amdgcn_wmma_f32_16x16x32_bf16(false, a1, false, bv1, (short)0, accv, false, false);
    accu = __builtin_amdgcn_wmma_f32_16x16x32_bf16(false, a0, false, bu0, (short)0, accu, false, false);
    accu = __builtin_amdgcn_wmma_f32_16x16x32_bf16(false, a1, false, bu1, (short)0, accu, false, false);

    // C/D layout: VGPR r, lane<16 -> (M=r, N=lane); lane>=16 -> (M=r+8, N=lane-16).
    const float bvl = bvp[h], bul = bup[h], wwl = Ww[h];
    float t[8];
#pragma unroll
    for (int r = 0; r < 8; ++r) {
        const float vv = tanhf(accv[r] + bvl);
        const float uu = 1.0f / (1.0f + __expf(-(accu[r] + bul)));
        t[r] = vv * uu * wwl;
    }
    // Reduce over the 16 h-columns of this wave (stays within each 16-lane half).
#pragma unroll
    for (int r = 0; r < 8; ++r) {
#pragma unroll
        for (int s = 1; s < 16; s <<= 1) t[r] += __shfl_xor(t[r], s, 32);
    }
    if (col == 0) {
#pragma unroll
        for (int r = 0; r < 8; ++r) atomicAdd(&part[r + half * 8], t[r]);
    }
    __syncthreads();
    if (tid < 16) att[n0 + tid] = part[tid] + bwp[0];
}

// ---------------------------------------------------------------------------
// Kernel 3: per-bag softmax over 64 contiguous logits. One 64-thread block/bag.
// ---------------------------------------------------------------------------
__global__ __launch_bounds__(64) void k_softmax(const float* __restrict__ att,
                                                float* __restrict__ w,
                                                float* __restrict__ out_attn) {
    __shared__ float red[64];
    const int b = blockIdx.x;
    const int t = threadIdx.x;
    const float a = att[b * SZB + t];
    red[t] = a;
    __syncthreads();
#pragma unroll
    for (int s = 32; s >= 1; s >>= 1) {
        if (t < s) red[t] = fmaxf(red[t], red[t + s]);
        __syncthreads();
    }
    const float mx = red[0];
    __syncthreads();
    const float e = __expf(a - mx);
    red[t] = e;
    __syncthreads();
#pragma unroll
    for (int s = 32; s >= 1; s >>= 1) {
        if (t < s) red[t] += red[t + s];
        __syncthreads();
    }
    const float sw = e / red[0];
    w[b * SZB + t]        = sw;
    out_attn[b * SZB + t] = sw;
}

// ---------------------------------------------------------------------------
// Kernel 4: bag[b,l,d] = sum_i w[b*SZB+i] * x[(b*SZB+i), l, d].
// One block per (b,l); 256 threads over D. Streams the second 1 GiB pass.
// ---------------------------------------------------------------------------
__global__ __launch_bounds__(256) void k_bagsum(const float* __restrict__ x,
                                                const float* __restrict__ w,
                                                float* __restrict__ out) {
    __shared__ float sw[SZB];
    const int b = blockIdx.x / LL;
    const int l = blockIdx.x % LL;
    const int d = threadIdx.x;
    if (threadIdx.x < SZB) sw[threadIdx.x] = w[b * SZB + threadIdx.x];
    __syncthreads();

    const float* xp = x + ((long long)b * SZB * LL + l) * DD + d;
    const long long stride = (long long)LL * DD;
    float acc = 0.0f;
#pragma unroll 4
    for (int i = 0; i < SZB; ++i) {
        if (i + 4 < SZB) __builtin_prefetch(xp + (long long)(i + 4) * stride, 0, 0);
        acc += sw[i] * xp[(long long)i * stride];
    }
    out[((long long)b * LL + l) * DD + d] = acc;
}

// ---------------------------------------------------------------------------
extern "C" void kernel_launch(void* const* d_in, const int* in_sizes, int n_in,
                              void* d_out, int out_size, void* d_ws, size_t ws_size,
                              hipStream_t stream) {
    (void)in_sizes; (void)n_in; (void)out_size; (void)ws_size;
    const float* x  = (const float*)d_in[0];   // inter_pre (N, L, 1, D)
    const float* Wv = (const float*)d_in[1];   // (H, L)
    const float* bv = (const float*)d_in[2];   // (H,)
    const float* Wu = (const float*)d_in[3];   // (H, L)
    const float* bu = (const float*)d_in[4];   // (H,)
    const float* Ww = (const float*)d_in[5];   // (1, H)
    const float* bw = (const float*)d_in[6];   // (1,)
    // d_in[7] = bags_size: uniform SZB per reference setup.

    float* out = (float*)d_out;                       // [B*L*D] bag out, then [N] softmax_attn
    float* m   = (float*)d_ws;                        // N*L floats
    float* att = m + (long long)NN * LL;              // N floats
    float* w   = att + NN;                            // N floats

    k_rowmax <<<(NN * LL) / 8, 256, 0, stream>>>(x, m);
    k_att    <<<NN / 16,       256, 0, stream>>>(m, Wv, bv, Wu, bu, Ww, bw, att);
    k_softmax<<<NB,             64, 0, stream>>>(att, w, out + (long long)NB * LL * DD);
    k_bagsum <<<NB * LL,       256, 0, stream>>>(x, w, out);
}
